// down_block_72885595013173
// MI455X (gfx1250) — matrix operand order repaired and checked by reading, not compiled, not run on hardware
//
#include <hip/hip_runtime.h>
#include <stdint.h>

#define B_      4
#define IN_CH   64
#define OUT_CH  128
#define N_HIGH  163842
#define N_LOW   40962
#define EPS_    1e-5f
#define SLOPE_  0.2f

typedef __bf16 v16bf __attribute__((ext_vector_type(16)));
typedef float  v8f   __attribute__((ext_vector_type(8)));

union Frag {
    uint4 u[2];
    v16bf v;
};

__device__ __forceinline__ uint16_t f2bf(float f) {
    union { float f; uint32_t u; } cv; cv.f = f;
    uint32_t u = cv.u;
    return (uint16_t)((u + 0x7FFFu + ((u >> 16) & 1u)) >> 16);  // RNE
}

// ---------------------------------------------------------------- weights -> bf16
__global__ __launch_bounds__(256) void k_cvt_w(const float* __restrict__ W1,
                                               const float* __restrict__ W2,
                                               uint16_t* __restrict__ Wb1,
                                               uint16_t* __restrict__ Wb2,
                                               int n1, int n2) {
    int t = blockIdx.x * 256 + threadIdx.x;
    if (t < n1)            Wb1[t]      = f2bf(W1[t]);
    else if (t < n1 + n2)  Wb2[t - n1] = f2bf(W2[t - n1]);
}

// ---------------------------------------------------------------- mean pool -> h0 bf16 (b,n,c)
__global__ __launch_bounds__(256) void k_pool(const float* __restrict__ x,
                                              const int* __restrict__ pidx,
                                              uint16_t* __restrict__ h0) {
    int slot = blockIdx.x * 4 + (threadIdx.x >> 6);   // (b,n) slot; 64 c-threads each
    int c    = threadIdx.x & 63;
    if (slot >= B_ * N_LOW) return;
    int b = slot / N_LOW;
    int n = slot % N_LOW;
    const float* xb = x + ((size_t)b * IN_CH + c) * N_HIGH;
    float s = 0.f;
#pragma unroll
    for (int j = 0; j < 7; ++j) s += xb[pidx[n * 7 + j]];
    h0[(size_t)slot * IN_CH + c] = f2bf(s * (1.0f / 7.0f));
}

// ---------------------------------------------------------------- one-ring conv via WMMA bf16
// hin: (b, n, CIN) bf16; Wb: (128, K) bf16 row-major; y: (b, n, 128) f32
// part: per-block deterministic GN partial sums [pb][8] = {sum,g0 sq,g0 ... sum,g3 sq,g3}
template <int CIN, int K, int ROWS>
__global__ __launch_bounds__(256) void k_conv(const uint16_t* __restrict__ hin,
                                              const int* __restrict__ neigh,
                                              const uint16_t* __restrict__ Wb,
                                              const float* __restrict__ bias,
                                              float* __restrict__ y,
                                              float* __restrict__ part) {
    constexpr int KP  = K + 8;        // padded LDS row stride (bf16), keeps 16B alignment, spreads banks
    constexpr int RT  = ROWS / 16;    // row tiles
    constexpr int WPR = 8 / RT;       // waves per row tile
    constexpr int TPW = RT;           // 16-col tiles per wave
    static_assert(TPW <= 4, "acc regs");

    __shared__ __align__(16) uint16_t sA[ROWS * KP];
    __shared__ float swred[8 * 8];    // [wave][8]

    const int b   = blockIdx.y;
    const int n0  = blockIdx.x * ROWS;
    const int tid = threadIdx.x;

    // ---- stage gathered A panel: ROWS x (7*CIN), 16B chunks
    constexpr int CHUNK = CIN / 8;            // 16B chunks per neighbor row
    constexpr int NCH   = ROWS * 7 * CHUNK;
    for (int t = tid; t < NCH; t += 256) {
        int r   = t / (7 * CHUNK);
        int rem = t % (7 * CHUNK);
        int j   = rem / CHUNK;
        int cc  = (rem % CHUNK) * 8;
        int n   = n0 + r; if (n >= N_LOW) n = N_LOW - 1;     // clamp; stores are guarded later
        int m   = neigh[n * 7 + j];
        const uint4* src = (const uint4*)(hin + ((size_t)b * N_LOW + m) * CIN + cc);
        uint4* dst = (uint4*)(&sA[r * KP + j * CIN + cc]);
        *dst = *src;
    }
    __syncthreads();

    const int wave = tid >> 5;
    const int lane = tid & 31;
    const int half = lane >> 4;
    const int l16  = lane & 15;
    const int rt   = wave / WPR;
    const int cw   = wave % WPR;

    v8f acc[TPW];
#pragma unroll
    for (int t = 0; t < TPW; ++t)
#pragma unroll
        for (int i = 0; i < 8; ++i) acc[t][i] = 0.f;

    const int arow = rt * 16 + l16;   // A-matrix M index (same for both lane halves)
#pragma unroll 2
    for (int kt = 0; kt < K / 32; ++kt) {
        // A frag: lanes 0-15 -> K {0..7,16..23}; lanes 16-31 -> K {8..15,24..31}
        Frag a;
        const uint16_t* ap = &sA[arow * KP + kt * 32 + half * 8];
        a.u[0] = *(const uint4*)(ap);
        a.u[1] = *(const uint4*)(ap + 16);
#pragma unroll
        for (int t = 0; t < TPW; ++t) {
            int cb = (cw * TPW + t) * 16;
            int o  = cb + l16;
            // B frag: lanes 0-15 -> K {0..7,16..23}; lanes 16-31 -> K {8..15,24..31}
            Frag bfr;
            const uint16_t* bp = Wb + (size_t)o * K + kt * 32 + half * 8;
            bfr.u[0] = *(const uint4*)(bp);
            bfr.u[1] = *(const uint4*)(bp + 16);
            acc[t] = __builtin_amdgcn_wmma_f32_16x16x32_bf16(
                false, a.v, false, bfr.v, (short)0, acc[t], false, false);
        }
    }

    // ---- epilogue: bias, store, deterministic GN partials
    float lsum[4] = {0.f, 0.f, 0.f, 0.f};
    float lsq[4]  = {0.f, 0.f, 0.f, 0.f};
#pragma unroll
    for (int t = 0; t < TPW; ++t) {
        int cb = (cw * TPW + t) * 16;
        int o  = cb + l16;
        int g  = cb >> 5;                       // 16-col tile lies inside one GN group
        float bv = bias[o];
#pragma unroll
        for (int r = 0; r < 8; ++r) {
            int n = n0 + rt * 16 + half * 8 + r;   // D layout: lanes0-15 M=r, lanes16-31 M=r+8
            if (n < N_LOW) {
                float v = acc[t][r] + bv;
                y[((size_t)b * N_LOW + n) * OUT_CH + o] = v;
                lsum[g] += v;
                lsq[g]  += v * v;
            }
        }
    }
    // wave tree-reduce (fixed order -> deterministic)
#pragma unroll
    for (int g = 0; g < 4; ++g) {
#pragma unroll
        for (int off = 16; off > 0; off >>= 1) {
            lsum[g] += __shfl_xor(lsum[g], off, 32);
            lsq[g]  += __shfl_xor(lsq[g],  off, 32);
        }
    }
    if (lane == 0) {
#pragma unroll
        for (int g = 0; g < 4; ++g) {
            swred[wave * 8 + 2 * g]     = lsum[g];
            swred[wave * 8 + 2 * g + 1] = lsq[g];
        }
    }
    __syncthreads();
    if (tid < 8) {
        float s = 0.f;
#pragma unroll
        for (int w = 0; w < 8; ++w) s += swred[w * 8 + tid];
        int pb = blockIdx.y * gridDim.x + blockIdx.x;
        part[(size_t)pb * 8 + tid] = s;
    }
}

// ---------------------------------------------------------------- fold block partials (fixed order)
__global__ void k_redstats(const float* __restrict__ part, float* __restrict__ stats, int nblk) {
    int t = threadIdx.x;           // 0..31 : b = t>>3, slot = t&7
    if (t >= 32) return;
    int b = t >> 3, i = t & 7;
    float s = 0.f;
    for (int blk = 0; blk < nblk; ++blk)
        s += part[((size_t)b * nblk + blk) * 8 + i];
    stats[t] = s;
}

// ---------------------------------------------------------------- GN + LeakyReLU -> bf16 (b,n,c)
__global__ __launch_bounds__(256) void k_norm1(const float* __restrict__ y,
                                               const float* __restrict__ stats,
                                               const float* __restrict__ gw,
                                               const float* __restrict__ gb,
                                               uint16_t* __restrict__ h1) {
    size_t idx = (size_t)blockIdx.x * 256 + threadIdx.x;
    size_t total = (size_t)B_ * N_LOW * OUT_CH;
    if (idx >= total) return;
    int c = (int)(idx & (OUT_CH - 1));
    int b = (int)(idx / ((size_t)N_LOW * OUT_CH));
    int g = c >> 5;
    const float cnt = 32.0f * (float)N_LOW;
    float mu  = stats[b * 8 + 2 * g] / cnt;
    float var = stats[b * 8 + 2 * g + 1] / cnt - mu * mu;
    float rs  = rsqrtf(var + EPS_);
    float v = (y[idx] - mu) * rs * gw[c] + gb[c];
    v = (v >= 0.f) ? v : SLOPE_ * v;
    h1[idx] = f2bf(v);
}

// ---------------------------------------------------------------- GN + LeakyReLU -> f32 out (b,c,n)
__global__ __launch_bounds__(256) void k_norm2(const float* __restrict__ y,
                                               const float* __restrict__ stats,
                                               const float* __restrict__ gw,
                                               const float* __restrict__ gb,
                                               float* __restrict__ out) {
    size_t idx = (size_t)blockIdx.x * 256 + threadIdx.x;   // idx = (b*128 + c)*N_LOW + n
    size_t total = (size_t)B_ * N_LOW * OUT_CH;
    if (idx >= total) return;
    int n = (int)(idx % N_LOW);
    int c = (int)((idx / N_LOW) % OUT_CH);
    int b = (int)(idx / ((size_t)N_LOW * OUT_CH));
    int g = c >> 5;
    const float cnt = 32.0f * (float)N_LOW;
    float mu  = stats[b * 8 + 2 * g] / cnt;
    float var = stats[b * 8 + 2 * g + 1] / cnt - mu * mu;
    float rs  = rsqrtf(var + EPS_);
    float v = (y[((size_t)b * N_LOW + n) * OUT_CH + c] - mu) * rs * gw[c] + gb[c];
    v = (v >= 0.f) ? v : SLOPE_ * v;
    out[idx] = v;
}

// ---------------------------------------------------------------- host
extern "C" void kernel_launch(void* const* d_in, const int* in_sizes, int n_in,
                              void* d_out, int out_size, void* d_ws, size_t ws_size,
                              hipStream_t stream) {
    (void)in_sizes; (void)n_in; (void)out_size; (void)ws_size;
    const float* x    = (const float*)d_in[0];
    const int*   pidx = (const int*)d_in[1];
    const int*   cidx = (const int*)d_in[2];
    const float* W1   = (const float*)d_in[3];
    const float* b1   = (const float*)d_in[4];
    const float* g1w  = (const float*)d_in[5];
    const float* g1b  = (const float*)d_in[6];
    const float* W2   = (const float*)d_in[7];
    const float* b2   = (const float*)d_in[8];
    const float* g2w  = (const float*)d_in[9];
    const float* g2b  = (const float*)d_in[10];
    float* out = (float*)d_out;

    char* ws = (char*)d_ws;
    auto alloc = [&](size_t bytes) {
        char* p = ws;
        ws += (bytes + 255) & ~(size_t)255;
        return p;
    };
    const int nb1 = (N_LOW + 63) / 64;   // conv1 blocks per batch
    const int nb2 = (N_LOW + 31) / 32;   // conv2 blocks per batch

    uint16_t* h0   = (uint16_t*)alloc((size_t)B_ * N_LOW * IN_CH  * 2);
    uint16_t* h1   = (uint16_t*)alloc((size_t)B_ * N_LOW * OUT_CH * 2);
    float*    y    = (float*)   alloc((size_t)B_ * N_LOW * OUT_CH * 4);   // reused both layers
    uint16_t* Wb1  = (uint16_t*)alloc((size_t)128 * 448 * 2);
    uint16_t* Wb2  = (uint16_t*)alloc((size_t)128 * 896 * 2);
    float*    prt1 = (float*)   alloc((size_t)B_ * nb1 * 8 * 4);
    float*    prt2 = (float*)   alloc((size_t)B_ * nb2 * 8 * 4);
    float*    st1  = (float*)   alloc(32 * 4);
    float*    st2  = (float*)   alloc(32 * 4);

    const int nw = 128 * 448 + 128 * 896;
    k_cvt_w<<<(nw + 255) / 256, 256, 0, stream>>>(W1, W2, Wb1, Wb2, 128 * 448, 128 * 896);

    k_pool<<<(B_ * N_LOW + 3) / 4, 256, 0, stream>>>(x, pidx, h0);

    dim3 gc1(nb1, B_);
    k_conv<IN_CH, 7 * IN_CH, 64><<<gc1, 256, 0, stream>>>(h0, cidx, Wb1, b1, y, prt1);
    k_redstats<<<1, 32, 0, stream>>>(prt1, st1, nb1);

    const size_t total = (size_t)B_ * N_LOW * OUT_CH;
    k_norm1<<<(unsigned)((total + 255) / 256), 256, 0, stream>>>(y, st1, g1w, g1b, h1);

    dim3 gc2(nb2, B_);
    k_conv<OUT_CH, 7 * OUT_CH, 32><<<gc2, 256, 0, stream>>>(h1, cidx, Wb2, b2, y, prt2);
    k_redstats<<<1, 32, 0, stream>>>(prt2, st2, nb2);

    k_norm2<<<(unsigned)((total + 255) / 256), 256, 0, stream>>>(y, st2, g2w, g2b, out);
}
